// AttentionReducer_78726750536347
// MI455X (gfx1250) — matrix operand order repaired and checked by reading, not compiled, not run on hardware
//
#include <hip/hip_runtime.h>
#include <hip/hip_bf16.h>
#include <cstdint>
#include <cstddef>

// ---------------------------------------------------------------------------
// Types for CDNA5 WMMA (wave32): D(16x16 f32) = A(16x32 bf16) * B(32x16 bf16) + C
// ---------------------------------------------------------------------------
typedef __bf16 bf16_t;
typedef __attribute__((ext_vector_type(16))) __bf16 v16bf;
typedef __attribute__((ext_vector_type(8)))  __bf16 v8bf;
typedef __attribute__((ext_vector_type(8)))  float  v8f;

#define WMMA_BF16(a, b, c) \
  __builtin_amdgcn_wmma_f32_16x16x32_bf16(false, (a), false, (b), (short)0, (c), false, false)

static __device__ __forceinline__ unsigned short f2bf(float f) {
  unsigned int x = __float_as_uint(f);
  unsigned int r = (x + 0x7fffu + ((x >> 16) & 1u)) >> 16;  // RNE
  return (unsigned short)r;
}
static __device__ __forceinline__ float bf2f(unsigned short h) {
  return __uint_as_float(((unsigned int)h) << 16);
}
static __device__ __forceinline__ v8f vzero8() {
  v8f z;
#pragma unroll
  for (int i = 0; i < 8; ++i) z[i] = 0.0f;
  return z;
}

// Model dims
#define MB    16
#define NSEQ  2048
#define MTOK  (MB * NSEQ)          // 32768
#define EMB   384
#define IN_D  1024
#define HEADS 8
#define DHEAD 48
#define INNER 384
#define WIN   64
#define NW    (NSEQ / WIN)         // 32
#define KWIN  (3 * WIN)            // 192
#define FFI   1024                 // FF_INNER
#define NEGI  (-1e30f)

// ---------------------------------------------------------------------------
// Weight swizzle: fp32 [K x N] row-major -> bf16 B-fragment tiles.
// Tile (kt, nt): 32x16, laid out as lane(0..31) x i(0..15) contiguous:
//   value(lane, i) = W[kt*32 + 8*(lane/16) + (i%8) + 16*(i/8)][nt*16 + (lane%16)]
// Tiles ordered nt-major so the GEMM k-loop walks contiguous 1KB tiles.
// ---------------------------------------------------------------------------
__global__ void k_swizzle_w(const float* __restrict__ src,
                            unsigned short* __restrict__ dst, int K, int N) {
  int idx = blockIdx.x * 256 + threadIdx.x;
  if (idx >= K * N) return;
  int k = idx / N, n = idx - k * N;
  int nt = n >> 4, l16 = n & 15;
  int kt = k >> 5, kk = k & 31;
  int half = (kk >> 3) & 1, i = (kk & 7) + ((kk >> 4) << 3);
  size_t off = ((size_t)nt * (size_t)(K >> 5) + (size_t)kt) * 512
             + (size_t)(half * 16 + l16) * 16 + (size_t)i;
  dst[off] = f2bf(src[(size_t)k * N + n]);
}

// ---------------------------------------------------------------------------
// xpos tables: cos/sin/scale, shape [2048][24]
// ---------------------------------------------------------------------------
__global__ void k_tables(float* __restrict__ cosb, float* __restrict__ sinb,
                         float* __restrict__ scb) {
  int idx = blockIdx.x * 256 + threadIdx.x;
  if (idx >= NSEQ * 24) return;
  int pos = idx / 24, i = idx - pos * 24;
  float invf = __powf(10000.0f, -(2.0f * (float)i) / 48.0f);
  float fr = (float)pos * invf;
  cosb[idx] = __cosf(fr);
  sinb[idx] = __sinf(fr);
  float sv = (2.0f * (float)i + 0.4f * 48.0f) / (1.4f * 48.0f);
  scb[idx] = __powf(sv, ((float)pos - 1024.0f) / 32.0f);
}

// h = x[:, :, :384]
__global__ void k_slice(const float* __restrict__ x, float* __restrict__ h) {
  int idx = blockIdx.x * 256 + threadIdx.x;
  if (idx >= MTOK * EMB) return;
  int row = idx / EMB, c = idx - row * EMB;
  h[idx] = x[(size_t)row * IN_D + c];
}

// ---------------------------------------------------------------------------
// LayerNorm: one block (128 thr) per row. BF16OUT: write bf16, else f32
// (f32 path is used in-place on d_out for the final LN).
// ---------------------------------------------------------------------------
template <bool BF16OUT>
__global__ __launch_bounds__(128) void k_ln(const float* __restrict__ src,
                                            const float* __restrict__ g,
                                            const float* __restrict__ b,
                                            unsigned short* __restrict__ dbf,
                                            float* __restrict__ df32, int K) {
  __shared__ float r1[128], r2[128];
  int row = blockIdx.x, tid = threadIdx.x;
  const float* s = src + (size_t)row * K;
  float sum = 0.f, sq = 0.f;
  for (int c = tid; c < K; c += 128) { float v = s[c]; sum += v; sq += v * v; }
  r1[tid] = sum; r2[tid] = sq;
  __syncthreads();
  for (int off = 64; off > 0; off >>= 1) {
    if (tid < off) { r1[tid] += r1[tid + off]; r2[tid] += r2[tid + off]; }
    __syncthreads();
  }
  float mean = r1[0] / (float)K;
  float var = r2[0] / (float)K - mean * mean;
  float rstd = rsqrtf(var + 1e-5f);
  for (int c = tid; c < K; c += 128) {
    float v = (s[c] - mean) * rstd * g[c] + b[c];
    if (BF16OUT) dbf[(size_t)row * K + c] = f2bf(v);
    else         df32[(size_t)row * K + c] = v;
  }
}

// ---------------------------------------------------------------------------
// Rotary (xpos) applied in place on bf16 qkv [MTOK x 1152]; q *= dhead^-0.5.
// One thread per (token, head, pair i<24), handles q and k.
// ---------------------------------------------------------------------------
__global__ void k_rotary(unsigned short* __restrict__ qkv,
                         const float* __restrict__ cosb,
                         const float* __restrict__ sinb,
                         const float* __restrict__ scb) {
  int idx = blockIdx.x * 256 + threadIdx.x;
  const int TOT = MTOK * HEADS * 24;
  if (idx >= TOT) return;
  int i = idx % 24;
  int t = idx / 24;
  int head = t % HEADS;
  int m = t / HEADS;
  int pos = m & (NSEQ - 1);
  float c = cosb[pos * 24 + i], s = sinb[pos * 24 + i], sc = scb[pos * 24 + i];
  size_t base = (size_t)m * (3 * INNER) + head * DHEAD;
  const float qs = 0.14433756729740643f;  // 48^-0.5
  float q0 = bf2f(qkv[base + i]), q1 = bf2f(qkv[base + i + 24]);
  qkv[base + i]      = f2bf((q0 * c - q1 * s) * sc * qs);
  qkv[base + i + 24] = f2bf((q1 * c + q0 * s) * sc * qs);
  size_t kb = base + INNER;
  float isc = 1.0f / sc;
  float k0 = bf2f(qkv[kb + i]), k1 = bf2f(qkv[kb + i + 24]);
  qkv[kb + i]      = f2bf((k0 * c - k1 * s) * isc);
  qkv[kb + i + 24] = f2bf((k1 * c + k0 * s) * isc);
}

// ---------------------------------------------------------------------------
// WMMA GEMM: C[MTOK x N] = A[MTOK x K](bf16) * Bsw(bf16, swizzled).
// One wave per 16x64 stripe (4 accumulators). 8 waves/block; nb = wid % (N/64) varies fastest, so a block may span M-tiles.
// Epilogues: 0=store bf16; 1=residual add into f32; 2=GEGLU (a*gelu(g), g
// tiles at nt + N/16 in a 2N-wide swizzled weight); 3=proj (+bias +x).
// ---------------------------------------------------------------------------
#define EP_BF16  0
#define EP_RESID 1
#define EP_GEGLU 2
#define EP_PROJ  3

template <int EPI>
__global__ __launch_bounds__(256) void k_gemm(
    const unsigned short* __restrict__ A, const unsigned short* __restrict__ Bsw,
    int K, int N, unsigned short* __restrict__ obf, float* __restrict__ of32,
    const float* __restrict__ bias, const float* __restrict__ xres) {
  const int KT = K >> 5;
  const int NB = N >> 6;
  int wid = (blockIdx.x * 256 + (int)threadIdx.x) >> 5;
  int lane = threadIdx.x & 31;
  int half = lane >> 4, l16 = lane & 15;
  int nb = wid % NB, mt = wid / NB;

  const unsigned short* Arow = A + (size_t)(mt * 16 + l16) * K;

  v8f acc[4], accg[4];
#pragma unroll
  for (int t = 0; t < 4; ++t) { acc[t] = vzero8(); if (EPI == EP_GEGLU) accg[t] = vzero8(); }

  for (int kt = 0; kt < KT; ++kt) {
    // A fragment: row m = mt*16 + l16; k = {kt*32+8h .. +7} and {kt*32+16+8h .. +7}
    v8bf a0 = *(const v8bf*)(Arow + kt * 32 + half * 8);
    v8bf a1 = *(const v8bf*)(Arow + kt * 32 + 16 + half * 8);
    v16bf af;
#pragma unroll
    for (int i = 0; i < 8; ++i) { af[i] = a0[i]; af[i + 8] = a1[i]; }
    if (kt + 1 < KT)
      __builtin_prefetch(Arow + (kt + 1) * 32, 0, 0);
#pragma unroll
    for (int t = 0; t < 4; ++t) {
      int nt = nb * 4 + t;
      v16bf bfr = *(const v16bf*)(Bsw + ((size_t)nt * KT + kt) * 512 + lane * 16);
      acc[t] = WMMA_BF16(af, bfr, acc[t]);
      if (EPI == EP_GEGLU) {
        v16bf bg = *(const v16bf*)(Bsw + ((size_t)(nt + (N >> 4)) * KT + kt) * 512 + lane * 16);
        accg[t] = WMMA_BF16(af, bg, accg[t]);
      }
    }
  }

#pragma unroll
  for (int t = 0; t < 4; ++t) {
    int n = (nb * 4 + t) * 16 + l16;
#pragma unroll
    for (int r = 0; r < 8; ++r) {
      size_t row = (size_t)mt * 16 + r + 8 * half;
      size_t o = row * (size_t)N + n;
      float v = acc[t][r];
      if (EPI == EP_BF16) {
        obf[o] = f2bf(v);
      } else if (EPI == EP_RESID) {
        of32[o] = v + of32[o];
      } else if (EPI == EP_PROJ) {
        of32[o] = v + bias[n] + xres[o];
      } else {  // EP_GEGLU: v = a, accg = g; out = a * gelu_exact(g)
        float gv = accg[t][r];
        float ge = 0.5f * gv * (1.0f + erff(gv * 0.70710678118654752f));
        obf[o] = f2bf(v * ge);
      }
    }
  }
}

// ---------------------------------------------------------------------------
// Local windowed attention. One block (128 thr = 4 waves) per (b, head, w).
// Wave ww owns q rows [ww*16, ww*16+16). K window and V^T are staged in LDS
// (zero-padded for invalid keys and for d=48 -> 64), so the WMMA loops are
// completely branch-free. The 24KB K-stage buffer is dead after sim and is
// reused for the P (softmax) matrix.
// ---------------------------------------------------------------------------
__global__ __launch_bounds__(128) void k_attn(const unsigned short* __restrict__ qkv,
                                              const int* __restrict__ mask,
                                              unsigned short* __restrict__ ctx) {
  __shared__ __align__(32) unsigned short vT[DHEAD * KWIN];  // [48][192] = 18KB
  __shared__ __align__(32) unsigned short kp[KWIN * 64];     // K [192][64] then P [64][192], 24KB

  const int bw = blockIdx.x;       // ((b*8 + hh)*32 + w)
  const int w = bw & 31;
  const int hh = (bw >> 5) & 7;
  const int b = bw >> 8;
  const int tid = threadIdx.x;
  const int lane = tid & 31, half = lane >> 4, l16 = lane & 15, ww = tid >> 5;

  // Stage K: kp[j*64 + dc] = k[g(j)][dc] (0 for invalid keys / dc >= 48).
  for (int idx = tid; idx < KWIN * 64; idx += 128) {
    int j = idx >> 6, dc = idx & 63;
    int g = (w - 1) * WIN + j;
    int gc = g < 0 ? 0 : (g > NSEQ - 1 ? NSEQ - 1 : g);
    bool valid = (g >= 0) && (g < NSEQ) && (dc < DHEAD) && (mask[b * NSEQ + gc] != 0);
    unsigned short val = qkv[((size_t)(b * NSEQ + gc)) * (3 * INNER) + INNER + hh * DHEAD +
                             (dc < DHEAD ? dc : 0)];
    kp[idx] = valid ? val : (unsigned short)0;
  }
  // Stage V^T: vT[dc][j] = v[g(j)][dc], zero for invalid keys.
  for (int idx = tid; idx < DHEAD * KWIN; idx += 128) {
    int dc = idx / KWIN, j = idx - dc * KWIN;
    int g = (w - 1) * WIN + j;
    int gc = g < 0 ? 0 : (g > NSEQ - 1 ? NSEQ - 1 : g);
    bool valid = (g >= 0) && (g < NSEQ) && (mask[b * NSEQ + gc] != 0);
    unsigned short val = qkv[((size_t)(b * NSEQ + gc)) * (3 * INNER) + 2 * INNER + hh * DHEAD + dc];
    vT[dc * KWIN + j] = valid ? val : (unsigned short)0;
  }
  __syncthreads();

  // ---- sim = Q K^T (branch-free; B frags contiguous from LDS) ----
  const unsigned short* qrow =
      qkv + ((size_t)(b * NSEQ + w * WIN + ww * 16 + l16)) * (3 * INNER) + hh * DHEAD;

  v8f s[12];
#pragma unroll
  for (int jt = 0; jt < 12; ++jt) s[jt] = vzero8();

#pragma unroll
  for (int ch = 0; ch < 2; ++ch) {
    v16bf aq;
    {
      v8bf a0 = *(const v8bf*)(qrow + ch * 32 + half * 8);  // k = ch*32 + 8h + i
#pragma unroll
      for (int i = 0; i < 8; ++i) aq[i] = a0[i];
      if (ch == 0) {
        v8bf a1 = *(const v8bf*)(qrow + 16 + half * 8);     // k = 16 + 8h + i
#pragma unroll
        for (int i = 0; i < 8; ++i) aq[i + 8] = a1[i];
      } else {
#pragma unroll
        for (int i = 0; i < 8; ++i) aq[i + 8] = (bf16_t)0.0f;  // k >= 48 pad
      }
    }
#pragma unroll
    for (int jt = 0; jt < 12; ++jt) {
      v8bf b0 = *(const v8bf*)(kp + (jt * 16 + l16) * 64 + ch * 32 + half * 8);
      v8bf b1 = *(const v8bf*)(kp + (jt * 16 + l16) * 64 + ch * 32 + 16 + half * 8);
      v16bf bk;
#pragma unroll
      for (int i = 0; i < 8; ++i) { bk[i] = b0[i]; bk[i + 8] = b1[i]; }
      s[jt] = WMMA_BF16(aq, bk, s[jt]);
    }
  }
  __syncthreads();  // kp (K stage) dead; about to be overwritten by P

  // ---- masked softmax over 192 keys, per row (r + 8*half) ----
  bool cv[12];
#pragma unroll
  for (int jt = 0; jt < 12; ++jt) {
    int j = jt * 16 + l16, g = (w - 1) * WIN + j;
    int gc = g < 0 ? 0 : (g > NSEQ - 1 ? NSEQ - 1 : g);
    cv[jt] = (g >= 0) && (g < NSEQ) && (mask[b * NSEQ + gc] != 0);
  }
#pragma unroll
  for (int r = 0; r < 8; ++r) {
    float mx = NEGI;
#pragma unroll
    for (int jt = 0; jt < 12; ++jt) mx = fmaxf(mx, cv[jt] ? s[jt][r] : NEGI);
    for (int d = 1; d < 16; d <<= 1) mx = fmaxf(mx, __shfl_xor(mx, d, 16));
    float sum = 0.f;
#pragma unroll
    for (int jt = 0; jt < 12; ++jt) {
      float v = cv[jt] ? __expf(s[jt][r] - mx) : 0.f;
      s[jt][r] = v;
      sum += v;
    }
    for (int d = 1; d < 16; d <<= 1) sum += __shfl_xor(sum, d, 16);
    float inv = 1.0f / sum;
#pragma unroll
    for (int jt = 0; jt < 12; ++jt)
      kp[(ww * 16 + r + 8 * half) * KWIN + jt * 16 + l16] = f2bf(s[jt][r] * inv);
  }
  // Each wave only reads back its own 16 P rows (written by itself): no
  // cross-wave barrier needed; per-wave LDS ordering handles the RAW.

  // ---- out = P V ----
  v8f o[3];
#pragma unroll
  for (int t = 0; t < 3; ++t) o[t] = vzero8();
  const unsigned short* prow = kp + (ww * 16 + l16) * KWIN;
#pragma unroll
  for (int ch = 0; ch < 6; ++ch) {
    v8bf a0 = *(const v8bf*)(prow + ch * 32 + half * 8);
    v8bf a1 = *(const v8bf*)(prow + ch * 32 + 16 + half * 8);
    v16bf ap;
#pragma unroll
    for (int i = 0; i < 8; ++i) { ap[i] = a0[i]; ap[i + 8] = a1[i]; }
#pragma unroll
    for (int t = 0; t < 3; ++t) {
      v8bf b0 = *(const v8bf*)(vT + (t * 16 + l16) * KWIN + ch * 32 + half * 8);
      v8bf b1 = *(const v8bf*)(vT + (t * 16 + l16) * KWIN + ch * 32 + 16 + half * 8);
      v16bf bv;
#pragma unroll
      for (int i = 0; i < 8; ++i) { bv[i] = b0[i]; bv[i + 8] = b1[i]; }
      o[t] = WMMA_BF16(ap, bv, o[t]);
    }
  }
#pragma unroll
  for (int t = 0; t < 3; ++t) {
    int dc = t * 16 + l16;
#pragma unroll
    for (int r = 0; r < 8; ++r) {
      size_t row = (size_t)b * NSEQ + w * WIN + ww * 16 + r + 8 * half;
      ctx[row * INNER + hh * DHEAD + dc] = f2bf(o[t][r]);
    }
  }
}

// ---------------------------------------------------------------------------
// Host orchestration.
// Input order assumes jax pytree (sorted dict keys) flattening:
// x, attention_mask, enc_ln_b, enc_ln_g,
//   per layer l: W1, W2, Wo, Wqkv, attn_ln_b, attn_ln_g, ff_ln_b, ff_ln_g,
// out_ln_b, out_ln_g, proj_W, proj_b
// ---------------------------------------------------------------------------
extern "C" void kernel_launch(void* const* d_in, const int* in_sizes, int n_in,
                              void* d_out, int out_size, void* d_ws, size_t ws_size,
                              hipStream_t stream) {
  (void)in_sizes; (void)n_in; (void)out_size; (void)ws_size;
  const float* x = (const float*)d_in[0];
  const int* mask = (const int*)d_in[1];
  const float* enc_ln_b = (const float*)d_in[2];
  const float* enc_ln_g = (const float*)d_in[3];
  const float *W1[3], *W2[3], *Wo[3], *Wqkv[3];
  const float *aln_b[3], *aln_g[3], *fln_b[3], *fln_g[3];
  for (int l = 0; l < 3; ++l) {
    int base = 4 + l * 8;
    W1[l]    = (const float*)d_in[base + 0];
    W2[l]    = (const float*)d_in[base + 1];
    Wo[l]    = (const float*)d_in[base + 2];
    Wqkv[l]  = (const float*)d_in[base + 3];
    aln_b[l] = (const float*)d_in[base + 4];
    aln_g[l] = (const float*)d_in[base + 5];
    fln_b[l] = (const float*)d_in[base + 6];
    fln_g[l] = (const float*)d_in[base + 7];
  }
  const float* out_ln_b = (const float*)d_in[28];
  const float* out_ln_g = (const float*)d_in[29];
  const float* projW = (const float*)d_in[30];
  const float* projb = (const float*)d_in[31];
  float* out = (float*)d_out;

  // Workspace carve (256B aligned)
  char* ws = (char*)d_ws;
  size_t off = 0;
  auto carve = [&](size_t bytes) -> char* {
    off = (off + 255) & ~(size_t)255;
    char* p = ws + off;
    off += bytes;
    return p;
  };
  unsigned short *qkvw[3], *wow[3], *w1w[3], *w2w[3];
  for (int l = 0; l < 3; ++l) {
    qkvw[l] = (unsigned short*)carve((size_t)EMB * 1152 * 2);
    wow[l]  = (unsigned short*)carve((size_t)INNER * EMB * 2);
    w1w[l]  = (unsigned short*)carve((size_t)EMB * 2048 * 2);
    w2w[l]  = (unsigned short*)carve((size_t)FFI * EMB * 2);
  }
  unsigned short* projw_s = (unsigned short*)carve((size_t)EMB * IN_D * 2);
  float* cosb = (float*)carve((size_t)NSEQ * 24 * 4);
  float* sinb = (float*)carve((size_t)NSEQ * 24 * 4);
  float* scb  = (float*)carve((size_t)NSEQ * 24 * 4);
  float* h = (float*)carve((size_t)MTOK * EMB * 4);
  unsigned short* abf  = (unsigned short*)carve((size_t)MTOK * EMB * 2);
  unsigned short* qkvb = (unsigned short*)carve((size_t)MTOK * 1152 * 2);
  unsigned short* ctxb = (unsigned short*)carve((size_t)MTOK * INNER * 2);
  unsigned short* gated = qkvb;  // alias: qkv dead by the FF phase

  auto swz = [&](const float* s, unsigned short* d, int K, int N) {
    int tot = K * N;
    k_swizzle_w<<<(tot + 255) / 256, 256, 0, stream>>>(s, d, K, N);
  };
  for (int l = 0; l < 3; ++l) {
    swz(Wqkv[l], qkvw[l], EMB, 1152);
    swz(Wo[l], wow[l], INNER, EMB);
    swz(W1[l], w1w[l], EMB, 2048);
    swz(W2[l], w2w[l], FFI, EMB);
  }
  swz(projW, projw_s, EMB, IN_D);

  k_tables<<<(NSEQ * 24 + 255) / 256, 256, 0, stream>>>(cosb, sinb, scb);
  k_slice<<<(MTOK * EMB + 255) / 256, 256, 0, stream>>>(x, h);

  auto gemm_grid = [](int N) { return (2048 * (N / 64)) / 8; };  // waves/8

  for (int l = 0; l < 3; ++l) {
    // LN -> QKV GEMM -> rotary
    k_ln<true><<<MTOK, 128, 0, stream>>>(h, aln_g[l], aln_b[l], abf, nullptr, EMB);
    k_gemm<EP_BF16><<<gemm_grid(1152), 256, 0, stream>>>(
        abf, qkvw[l], EMB, 1152, qkvb, nullptr, nullptr, nullptr);
    k_rotary<<<(MTOK * HEADS * 24 + 255) / 256, 256, 0, stream>>>(qkvb, cosb, sinb, scb);
    // attention
    k_attn<<<MB * HEADS * NW, 128, 0, stream>>>(qkvb, mask, ctxb);
    // Wo GEMM + residual into h
    k_gemm<EP_RESID><<<gemm_grid(EMB), 256, 0, stream>>>(
        ctxb, wow[l], INNER, EMB, nullptr, h, nullptr, nullptr);
    // FF: LN -> GEGLU GEMM -> W2 GEMM + residual
    k_ln<true><<<MTOK, 128, 0, stream>>>(h, fln_g[l], fln_b[l], abf, nullptr, EMB);
    k_gemm<EP_GEGLU><<<gemm_grid(FFI), 256, 0, stream>>>(
        abf, w1w[l], EMB, FFI, gated, nullptr, nullptr, nullptr);
    k_gemm<EP_RESID><<<gemm_grid(EMB), 256, 0, stream>>>(
        gated, w2w[l], FFI, EMB, nullptr, h, nullptr, nullptr);
  }

  // Final: LN(enc) -> proj GEMM (+bias +x) -> LN(out) in place on d_out
  k_ln<true><<<MTOK, 128, 0, stream>>>(h, enc_ln_g, enc_ln_b, abf, nullptr, EMB);
  k_gemm<EP_PROJ><<<gemm_grid(IN_D), 256, 0, stream>>>(
      abf, projw_s, EMB, IN_D, nullptr, out, projb, x);
  k_ln<false><<<MTOK, 128, 0, stream>>>(out, out_ln_g, out_ln_b, nullptr, out, IN_D);
}
